// GridSelfAttention_33560874451072
// MI455X (gfx1250) — matrix operand order repaired and checked by reading, not compiled, not run on hardware
//
#include <hip/hip_runtime.h>

#define USE_TDM 1

typedef __bf16 bf16;
typedef __attribute__((ext_vector_type(16))) __bf16 bf16x16;
typedef __attribute__((ext_vector_type(8)))  __bf16 bf16x8;
typedef __attribute__((ext_vector_type(8)))  float  f32x8;
typedef __attribute__((ext_vector_type(4)))  unsigned int u32x4;
typedef __attribute__((ext_vector_type(8)))  int i32x8;
typedef __attribute__((ext_vector_type(4)))  int i32x4;

__device__ __forceinline__ bf16x16 join16(bf16x8 lo, bf16x8 hi) {
  union { bf16x16 v; bf16x8 h[2]; } u;
  u.h[0] = lo; u.h[1] = hi;
  return u.v;
}

__device__ __forceinline__ f32x8 wmma_bf16(bf16x16 a, bf16x16 b, f32x8 c) {
  return __builtin_amdgcn_wmma_f32_16x16x32_bf16(false, a, false, b, (short)0, c,
                                                 false, false);
}

__device__ __forceinline__ float gelu_tanh(float x) {
  const float x3 = x * x * x;
  return 0.5f * x * (1.f + tanhf(0.7978845608028654f * (x + 0.044715f * x3)));
}

// ------------------------------------------------------------------ TDM helpers
// 2-D tile load descriptor (D# groups 0/1 per CDNA5 ISA).
// Tile: tile1 rows x tile0 elements, element = 2 bytes (bf16).
// LDS padding: +4 DWORDs after every 16 DWORDs -> 80-byte LDS row stride.
struct TdmDesc {
  u32x4 g0;
  i32x8 g1;
};

__device__ __forceinline__ TdmDesc make_desc2d(const void* gaddr, unsigned lds_addr,
                                               unsigned tile0, unsigned tile1,
                                               unsigned long tensor0,
                                               unsigned long tensor1,
                                               unsigned long stride0) {
  const unsigned long long ga = (unsigned long long)gaddr;
  TdmDesc d;
  d.g0[0] = 1u;  // count=1, is_restore=0, gather off
  d.g0[1] = lds_addr;
  d.g0[2] = (unsigned)(ga & 0xffffffffu);
  d.g0[3] = (unsigned)((ga >> 32) & 0x1ffffffu) | (2u << 30);  // type=2 ("image")
  // data_size=1(2B), pad_enable=1, pad_interval=3(16 DW), pad_amount=3(4 DW)
  d.g1[0] = (int)((1u << 16) | (1u << 20) | (3u << 22) | (3u << 25));
  d.g1[1] = (int)((unsigned)(tensor0 & 0xffffu) << 16);  // abar=0 | tensor_dim0[15:0]
  d.g1[2] = (int)(((tensor0 >> 16) & 0xffffu) | ((tensor1 & 0xffffu) << 16));
  d.g1[3] = (int)(((tensor1 >> 16) & 0xffffu) | ((tile0 & 0xffffu) << 16));
  d.g1[4] = (int)(tile1 & 0xffffu);  // tile_dim1 | tile_dim2=0
  d.g1[5] = (int)(stride0 & 0xffffffffu);
  d.g1[6] = (int)((stride0 >> 32) & 0xffffu);  // | tensor_dim1_stride[15:0]=0
  d.g1[7] = 0;
  return d;
}

__device__ __forceinline__ void tdm_issue(TdmDesc d) {
  const i32x4 z4 = {};
#if __clang_major__ >= 23
  const i32x8 z8 = {};
  __builtin_amdgcn_tensor_load_to_lds(d.g0, d.g1, z4, z4, z8, 0);
#else
  __builtin_amdgcn_tensor_load_to_lds(d.g0, d.g1, z4, z4, 0);
#endif
}

__device__ __forceinline__ unsigned lds_offset_of(const void* p) {
  // LDS aperture: low 32 bits of a flat shared address are the LDS byte offset
  return (unsigned)(unsigned long long)p;
}

// ---------------------------------------------------------------- weights cvt
// converts fp32 W[K][N] into bf16 W^T[N][K] (GEMM consumes B transposed)
__global__ __launch_bounds__(256)
void cvt_f32_to_bf16_T(const float* __restrict__ s, bf16* __restrict__ d,
                       const int N, const int K) {
  const int i = blockIdx.x * 256 + threadIdx.x;
  if (i < N * K) {
    const int k = i / N, n = i - k * N;
    d[(long)n * K + k] = (bf16)s[i];
  }
}

// --------------------------------------------------------- dual LayerNorm (C=512)
__global__ __launch_bounds__(256)
void ln_dual_kernel(const float* __restrict__ x,
                    const float* __restrict__ g1, const float* __restrict__ b1,
                    const float* __restrict__ g2, const float* __restrict__ b2,
                    bf16* __restrict__ o1, bf16* __restrict__ o2) {
  const int wave = threadIdx.x >> 5, lane = threadIdx.x & 31;
  const long row = (long)blockIdx.x * 8 + wave;
  const float* xr = x + row * 512;
  float v[16];
  float s = 0.f;
#pragma unroll
  for (int i = 0; i < 4; ++i) {
    const float4 f = *(const float4*)(xr + i * 128 + lane * 4);
    v[4 * i + 0] = f.x; v[4 * i + 1] = f.y; v[4 * i + 2] = f.z; v[4 * i + 3] = f.w;
    s += f.x + f.y + f.z + f.w;
  }
#pragma unroll
  for (int m = 16; m >= 1; m >>= 1) s += __shfl_xor(s, m, 32);
  const float mean = s * (1.f / 512.f);
  float q = 0.f;
#pragma unroll
  for (int i = 0; i < 16; ++i) { const float d = v[i] - mean; q += d * d; }
#pragma unroll
  for (int m = 16; m >= 1; m >>= 1) q += __shfl_xor(q, m, 32);
  const float rstd = rsqrtf(q * (1.f / 512.f) + 1e-5f);
#pragma unroll
  for (int i = 0; i < 16; ++i) {
    const int c = (i >> 2) * 128 + lane * 4 + (i & 3);
    const float nv = (v[i] - mean) * rstd;
    o1[row * 512 + c] = (bf16)(nv * g1[c] + b1[c]);
    if (o2) o2[row * 512 + c] = (bf16)(nv * g2[c] + b2[c]);
  }
}

// -------------------------------------------------- tiled bf16 WMMA GEMM
// C[M,N] = A[M,K] @ B[K,N], with B supplied TRANSPOSED: Bt[N][K] row-major.
// 128x128x32 tiles, 256 threads (8 waves, 2x4 microtiles of 16x16 each),
// double-buffered LDS fed by the Tensor Data Mover (wave 0 issues, all wait).
// EPI 0: out = bf16(acc + bias)
// EPI 1: out = bf16(gelu(acc + bias))
// EPI 2: out(f32) = resid + gscale[col] * (acc + bias)
template <int EPI>
__global__ __launch_bounds__(256)
void gemm_bf16_kernel(const bf16* __restrict__ A, const bf16* __restrict__ Bt,
                      const float* __restrict__ bias, const float* __restrict__ gscale,
                      const float* __restrict__ resid, void* __restrict__ outp,
                      const int N, const int K) {
  __shared__ alignas(16) bf16 As[2][128][40];  // [m][k], 80B row stride (TDM pad)
  __shared__ alignas(16) bf16 Bs[2][128][40];  // [n][k]
  const int tid = threadIdx.x;
  const int wave = tid >> 5, lane = tid & 31;
  const int wm = wave & 3, wn = wave >> 2;  // 4x2 wave grid: 32x64 per wave
  const int mlane = lane & 15, half = lane >> 4;
  const long m0 = (long)blockIdx.x * 128;
  const int n0 = blockIdx.y * 128;

  f32x8 acc[2][4] = {};

#if USE_TDM
  const unsigned ldsA0 = lds_offset_of(&As[0][0][0]);
  const unsigned ldsA1 = lds_offset_of(&As[1][0][0]);
  const unsigned ldsB0 = lds_offset_of(&Bs[0][0][0]);
  const unsigned ldsB1 = lds_offset_of(&Bs[1][0][0]);

  auto issue_tiles = [&](int k0, int buf) {
    if (wave == 0) {
      tdm_issue(make_desc2d(A + m0 * K + k0, buf ? ldsA1 : ldsA0, 32, 128,
                            (unsigned long)K, 65536ul, (unsigned long)K));
      tdm_issue(make_desc2d(Bt + (long)n0 * K + k0, buf ? ldsB1 : ldsB0, 32, 128,
                            (unsigned long)K, (unsigned long)N, (unsigned long)K));
    }
  };
  issue_tiles(0, 0);
#else
  const int srow = tid >> 1, scol = (tid & 1) * 16;  // 128 rows x 32 cols staging
  const bf16* ag = A + (m0 + srow) * K + scol;
  const bf16* bg = Bt + (long)(n0 + srow) * K + scol;
#endif

  for (int k0 = 0; k0 < K; k0 += 32) {
    const int buf = (k0 >> 5) & 1;
#if USE_TDM
    if (wave == 0) __builtin_amdgcn_s_wait_tensorcnt(0);
    __syncthreads();
    if (k0 + 32 < K) issue_tiles(k0 + 32, buf ^ 1);
#else
    __syncthreads();
    *(bf16x8*)&As[buf][srow][scol] = *(const bf16x8*)(ag + k0);
    *(bf16x8*)&As[buf][srow][scol + 8] = *(const bf16x8*)(ag + k0 + 8);
    *(bf16x8*)&Bs[buf][srow][scol] = *(const bf16x8*)(bg + k0);
    *(bf16x8*)&Bs[buf][srow][scol + 8] = *(const bf16x8*)(bg + k0 + 8);
    __syncthreads();
#endif
    // A fragment: lane(m=mlane): elems 0..7 -> K = half*8+i, 8..15 -> K = 16+half*8+i
    bf16x16 af[2];
#pragma unroll
    for (int i = 0; i < 2; ++i) {
      const bf16* p = &As[buf][wm * 32 + i * 16 + mlane][half * 8];
      af[i] = join16(*(const bf16x8*)p, *(const bf16x8*)(p + 16));
    }
    // B fragment: lane(n=mlane): elems 0..7 -> K = half*8+i, 8..15 -> K = 16+half*8+i
#pragma unroll
    for (int j = 0; j < 4; ++j) {
      const bf16* p = &Bs[buf][wn * 64 + j * 16 + mlane][half * 8];
      const bf16x16 bfj = join16(*(const bf16x8*)p, *(const bf16x8*)(p + 16));
#pragma unroll
      for (int i = 0; i < 2; ++i) acc[i][j] = wmma_bf16(af[i], bfj, acc[i][j]);
    }
    __syncthreads();
  }

  // epilogue; C layout: col = mlane, row = r + 8*half (per accumulator VGPR r)
#pragma unroll
  for (int j = 0; j < 4; ++j) {
    const int cn = n0 + wn * 64 + j * 16 + mlane;
    const float bs = bias[cn];
    const float gm = (EPI == 2) ? gscale[cn] : 0.f;
#pragma unroll
    for (int i = 0; i < 2; ++i) {
#pragma unroll
      for (int r = 0; r < 8; ++r) {
        const long rm = m0 + wm * 32 + i * 16 + half * 8 + r;
        const float vv = acc[i][j][r] + bs;
        if (EPI == 0)
          ((bf16*)outp)[rm * N + cn] = (bf16)vv;
        else if (EPI == 1)
          ((bf16*)outp)[rm * N + cn] = (bf16)gelu_tanh(vv);
        else
          ((float*)outp)[rm * N + cn] = resid[rm * N + cn] + gm * vv;
      }
    }
  }
}

// ----------------------------------------------------- windowed attention
// grid = (1024 windows, 8 heads), block = 128 (4 waves, one 16-query tile each)
// qb: bf16 [65536,512]; kvb: bf16 [65536,1024] (k = [0:512), v = [512:1024))
__global__ __launch_bounds__(128)
void attn_kernel(const bf16* __restrict__ qb, const bf16* __restrict__ kvb,
                 bf16* __restrict__ ao) {
  __shared__ alignas(16) bf16 VT[64][200];     // V transposed: [d][key]
  __shared__ alignas(16) bf16 AT[4][16][192];  // per-wave attn weights
  const int wgid = blockIdx.x;
  const int h = blockIdx.y;
  const int seq = wgid >> 7, w = wgid & 127;  // s = 128 windows/sequence
  const long seqbase = (long)seq * 8192;
  const int tid = threadIdx.x, wave = tid >> 5, lane = tid & 31;
  const int mlane = lane & 15, half = lane >> 4;

  auto keyrow = [&](int j) -> long {  // j in [0,192): windows (w-1,w,w+1) clipped
    int wn = w - 1 + (j >> 6);
    wn = wn < 0 ? 0 : (wn > 127 ? 127 : wn);
    return seqbase + (long)wn * 64 + (j & 63);
  };

  for (int idx = tid; idx < 192 * 64; idx += 128) {
    const int j = idx >> 6, d = idx & 63;
    VT[d][j] = kvb[keyrow(j) * 1024 + 512 + h * 64 + d];
  }
  __syncthreads();

  const long qrow0 = seqbase + (long)w * 64 + wave * 16;

  bf16x16 qf[2];
  {
    const bf16* qp = qb + (qrow0 + mlane) * 512 + h * 64;
#pragma unroll
    for (int ks = 0; ks < 2; ++ks) {
      const bf16* p = qp + ks * 32 + half * 8;
      qf[ks] = join16(*(const bf16x8*)p, *(const bf16x8*)(p + 16));
    }
  }

  f32x8 sacc[12] = {};
#pragma unroll
  for (int nt = 0; nt < 12; ++nt) {
    const bf16* kp = kvb + keyrow(nt * 16 + mlane) * 1024 + h * 64;
#pragma unroll
    for (int ks = 0; ks < 2; ++ks) {
      const bf16* p = kp + ks * 32 + half * 8;
      const bf16x16 kf = join16(*(const bf16x8*)p, *(const bf16x8*)(p + 16));
      sacc[nt] = wmma_bf16(qf[ks], kf, sacc[nt]);
    }
  }

  const bool mL = (w == 0), mR = (w == 127);
  float rmax[8], rsum[8];
#pragma unroll
  for (int r = 0; r < 8; ++r) rmax[r] = -3.0e38f;
#pragma unroll
  for (int nt = 0; nt < 12; ++nt) {
    const bool masked = (mL && nt < 4) || (mR && nt >= 8);
#pragma unroll
    for (int r = 0; r < 8; ++r) {
      const float vv = masked ? -3.0e38f : sacc[nt][r] * 0.125f;
      sacc[nt][r] = vv;
      rmax[r] = fmaxf(rmax[r], vv);
    }
  }
#pragma unroll
  for (int r = 0; r < 8; ++r) {
#pragma unroll
    for (int m = 8; m >= 1; m >>= 1)
      rmax[r] = fmaxf(rmax[r], __shfl_xor(rmax[r], m, 32));
    rsum[r] = 0.f;
  }
#pragma unroll
  for (int nt = 0; nt < 12; ++nt)
#pragma unroll
    for (int r = 0; r < 8; ++r) {
      const float p = __expf(sacc[nt][r] - rmax[r]);
      sacc[nt][r] = p;
      rsum[r] += p;
    }
#pragma unroll
  for (int r = 0; r < 8; ++r) {
#pragma unroll
    for (int m = 8; m >= 1; m >>= 1) rsum[r] += __shfl_xor(rsum[r], m, 32);
    rsum[r] = 1.f / rsum[r];
  }
#pragma unroll
  for (int nt = 0; nt < 12; ++nt)
#pragma unroll
    for (int r = 0; r < 8; ++r)
      AT[wave][r + 8 * half][nt * 16 + mlane] = (bf16)(sacc[nt][r] * rsum[r]);
  __syncthreads();

  f32x8 oacc[4] = {};
#pragma unroll
  for (int ks = 0; ks < 6; ++ks) {
    const bf16* ap = &AT[wave][mlane][ks * 32 + half * 8];
    const bf16x16 afr = join16(*(const bf16x8*)ap, *(const bf16x8*)(ap + 16));
#pragma unroll
    for (int ot = 0; ot < 4; ++ot) {
      const bf16* vp = &VT[ot * 16 + mlane][ks * 32 + half * 8];
      const bf16x16 vfr = join16(*(const bf16x8*)vp, *(const bf16x8*)(vp + 16));
      oacc[ot] = wmma_bf16(afr, vfr, oacc[ot]);
    }
  }
#pragma unroll
  for (int ot = 0; ot < 4; ++ot) {
    const int dcol = h * 64 + ot * 16 + mlane;
#pragma unroll
    for (int r = 0; r < 8; ++r)
      ao[(qrow0 + r + 8 * half) * 512 + dcol] = (bf16)(oacc[ot][r]);
  }
}

// -------------------------------------------------------------------- driver
extern "C" void kernel_launch(void* const* d_in, const int* in_sizes, int n_in,
                              void* d_out, int out_size, void* d_ws, size_t ws_size,
                              hipStream_t stream) {
  (void)in_sizes; (void)n_in; (void)out_size; (void)ws_size;
  const float* x         = (const float*)d_in[0];
  const float* ln_q_g    = (const float*)d_in[1];
  const float* ln_q_b    = (const float*)d_in[2];
  const float* ln_kv_g   = (const float*)d_in[3];
  const float* ln_kv_b   = (const float*)d_in[4];
  const float* W_kv      = (const float*)d_in[5];
  const float* b_kv      = (const float*)d_in[6];
  const float* W_o       = (const float*)d_in[7];
  const float* b_o       = (const float*)d_in[8];
  const float* gamma     = (const float*)d_in[9];
  const float* ln_m_g    = (const float*)d_in[10];
  const float* ln_m_b    = (const float*)d_in[11];
  const float* W_emb     = (const float*)d_in[12];
  const float* b_emb     = (const float*)d_in[13];
  const float* W1        = (const float*)d_in[14];
  const float* b1        = (const float*)d_in[15];
  const float* W2        = (const float*)d_in[16];
  const float* b2        = (const float*)d_in[17];
  const float* gamma_mlp = (const float*)d_in[18];
  float* out = (float*)d_out;

  char* ws = (char*)d_ws;
  // N = 65536 tokens, C = 512
  bf16* kvb   = (bf16*)(ws);                         // [N,1024] (also gelu(h1))
  bf16* qb    = (bf16*)(ws + 134217728);             // [N,512]  (also ln_m)
  bf16* aob   = (bf16*)(ws + 134217728 + 67108864);  // [N,512]  (ln_kv/attn/emb)
  bf16* wkvT  = (bf16*)(ws + 268435456);             // [1024,512]
  bf16* woT   = wkvT + 1024 * 512;                   // [512,512]
  bf16* wembT = woT + 512 * 512;                     // [512,512]
  bf16* w1T   = wembT + 512 * 512;                   // [1024,512]
  bf16* w2T   = w1T + 1024 * 512;                    // [512,1024]

  // 0) weights fp32 -> bf16, transposed to [N][K] for the GEMM B operand
  cvt_f32_to_bf16_T<<<2048, 256, 0, stream>>>(W_kv, wkvT, 1024, 512);
  cvt_f32_to_bf16_T<<<1024, 256, 0, stream>>>(W_o, woT, 512, 512);
  cvt_f32_to_bf16_T<<<1024, 256, 0, stream>>>(W_emb, wembT, 512, 512);
  cvt_f32_to_bf16_T<<<2048, 256, 0, stream>>>(W1, w1T, 1024, 512);
  cvt_f32_to_bf16_T<<<2048, 256, 0, stream>>>(W2, w2T, 512, 1024);

  // 1) q = LN_q(x), ln_kv = LN_kv(x)
  ln_dual_kernel<<<8192, 256, 0, stream>>>(x, ln_q_g, ln_q_b, ln_kv_g, ln_kv_b, qb, aob);
  // 2) kv = ln_kv @ W_kv + b_kv
  gemm_bf16_kernel<0><<<dim3(512, 8), 256, 0, stream>>>(aob, wkvT, b_kv, nullptr,
                                                        nullptr, (void*)kvb, 1024, 512);
  // 3) windowed attention
  attn_kernel<<<dim3(1024, 8), 128, 0, stream>>>(qb, kvb, aob);
  // 4) x1 = x + gamma * (attn @ W_o + b_o)      (fp32 running x in d_out)
  gemm_bf16_kernel<2><<<dim3(512, 4), 256, 0, stream>>>(aob, woT, b_o, gamma, x,
                                                        (void*)out, 512, 512);
  // 5) ln_m(x1)
  ln_dual_kernel<<<8192, 256, 0, stream>>>(out, ln_m_g, ln_m_b, ln_m_g, ln_m_b, qb,
                                           nullptr);
  // 6) h0 = ln_m @ W_emb + b_emb
  gemm_bf16_kernel<0><<<dim3(512, 4), 256, 0, stream>>>(qb, wembT, b_emb, nullptr,
                                                        nullptr, (void*)aob, 512, 512);
  // 7) h1 = gelu(h0 @ W1 + b1)
  gemm_bf16_kernel<1><<<dim3(512, 8), 256, 0, stream>>>(aob, w1T, b1, nullptr, nullptr,
                                                        (void*)kvb, 1024, 512);
  // 8) x2 = x1 + gamma_mlp * (h1 @ W2 + b2)
  gemm_bf16_kernel<2><<<dim3(512, 4), 256, 0, stream>>>(kvb, w2T, b2, gamma_mlp, out,
                                                        (void*)out, 512, 1024);
}
